// RouterPFSelfAttention_2989297238604
// MI455X (gfx1250) — matrix operand order repaired and checked by reading, not compiled, not run on hardware
//
#include <hip/hip_runtime.h>
#include <math.h>

typedef __attribute__((ext_vector_type(16))) _Float16 v16h;
typedef __attribute__((ext_vector_type(8)))  _Float16 v8h;
typedef __attribute__((ext_vector_type(8)))  float    v8f;
typedef __attribute__((ext_vector_type(4)))  unsigned int uint32x4;
typedef __attribute__((ext_vector_type(8)))  int          int32x8;
typedef __attribute__((ext_vector_type(4)))  int          int32x4;

#define BATCH  8
#define SEQ    2048
#define EMB    768
#define QTILE  128
#define KTILE  128
#define KPITCH (EMB + 8)      // pad to break LDS bank conflicts
#define NKSEG  (EMB / 32)     // 24 WMMA k-steps of 32

union HF16 { v16h v; v8h h[2]; };

// Kernel 1: v' = x @ W^T  -> [B*S, 2] in workspace (one wave per row)
__global__ __launch_bounds__(256)
void vproj_kernel(const float* __restrict__ x, const float* __restrict__ W,
                  float* __restrict__ vws) {
  int row  = blockIdx.x * 8 + (threadIdx.x >> 5);
  int lane = threadIdx.x & 31;
  const float* xr = x + (size_t)row * EMB;
  float s0 = 0.f, s1 = 0.f;
  for (int k = lane; k < EMB; k += 32) {
    float xv = xr[k];
    s0 += xv * W[k];
    s1 += xv * W[EMB + k];
  }
#pragma unroll
  for (int m = 16; m >= 1; m >>= 1) {
    s0 += __shfl_xor(s0, m, 32);
    s1 += __shfl_xor(s1, m, 32);
  }
  if (lane == 0) {
    vws[2 * row]     = s0;
    vws[2 * row + 1] = s1;
  }
}

// Kernel 2: flash attention with 2-dim values.
// Block = (batch b, 128-query tile). 8 waves, each owns a 16-row M tile.
__global__ __launch_bounds__(256)
void flash_attn_kernel(const float* __restrict__ x, const float* __restrict__ vws,
                       const float* __restrict__ bias, float* __restrict__ out) {
  __shared__ _Float16 ldsK[KTILE * KPITCH];   // ~194 KB (f16 K tile)
  __shared__ float2   ldsV[SEQ];              // 16 KB  (projected values)

  const int tid  = threadIdx.x;
  const int wave = tid >> 5;
  const int lane = tid & 31;
  const int b    = blockIdx.y;
  const int qbase = blockIdx.x * QTILE + wave * 16;
  const float* xb = x + (size_t)b * SEQ * EMB;
  const float qscale = 0.036084391824351615f;  // 1/sqrt(768)

  // --- Stage this batch's projected values into LDS via the Tensor Data Mover.
  // 2D descriptor: 4096 x 1 tile of 4-byte elements (16 KB), wave 0 issues,
  // TENSORcnt tracks completion, block barrier publishes to all waves.
  if (wave == 0) {
    unsigned long long gaddr = (unsigned long long)(const void*)(vws + (size_t)b * SEQ * 2);
    unsigned int ldsoff = (unsigned int)(size_t)(void*)&ldsV[0];
    uint32x4 g0;
    g0[0] = 1u;                                    // count = 1 valid descriptor
    g0[1] = ldsoff;                                // lds_addr (bytes)
    g0[2] = (unsigned int)(gaddr & 0xffffffffu);   // global_addr[31:0]
    g0[3] = (unsigned int)((gaddr >> 32) & 0x1ffffffu) | (2u << 30); // addr[56:32] | type=2
    int32x8 g1;
    g1[0] = 0x00020000;            // workgroup_mask=0, data_size=2 (4 bytes)
    g1[1] = (int)(4096u << 16);    // tensor_dim0 = 4096 (bits[63:48])
    g1[2] = (int)(1u << 16);       // tensor_dim1 = 1    (bits[111:80])
    g1[3] = (int)(4096u << 16);    // tile_dim0 = 4096   (bits[127:112])
    g1[4] = 1;                     // tile_dim1 = 1
    g1[5] = 4096;                  // tensor_dim0_stride
    g1[6] = 0;
    g1[7] = 0;
    int32x4 z4 = {0, 0, 0, 0};
    int32x8 z8 = {0, 0, 0, 0, 0, 0, 0, 0};
    __builtin_amdgcn_tensor_load_to_lds(g0, g1, z4, z4, z8, 0);
    __builtin_amdgcn_s_wait_tensorcnt(0);
  }
  __syncthreads();

  // Load Q fragments direct from global in this kernel's contiguous-k A order (same k order as B), fp32->f16.
  // lanes 0-15: M=lane, K = 0..15 ; lanes 16-31: M=lane-16, K = 16..31
  const int arow = lane & 15;
  const int akb  = (lane >> 4) * 16;
  const float* qrow = xb + (size_t)(qbase + arow) * EMB;
  v16h qf[NKSEG];
#pragma unroll
  for (int ks = 0; ks < NKSEG; ++ks) {
    const float* p = qrow + ks * 32 + akb;
    v16h q;
#pragma unroll
    for (int j = 0; j < 8; ++j) q[j]     = (_Float16)(p[j]      * qscale);
#pragma unroll
    for (int j = 0; j < 8; ++j) q[8 + j] = (_Float16)(p[8 + j] * qscale);
    qf[ks] = q;
  }

  // Online-softmax state (C/D layout: VGPR r -> row M=r / M=r+8).
  float runm[8], lsum[8], acc0[8], acc1[8];
#pragma unroll
  for (int r = 0; r < 8; ++r) { runm[r] = -INFINITY; lsum[r] = 0.f; acc0[r] = 0.f; acc1[r] = 0.f; }

  const int bcol = lane & 15;          // B-matrix column (= key row in tile)
  const int bk16 = (lane >> 4) * 16;   // lanes 16-31 hold K=16..31 of the 32-slice

  for (int kb = 0; kb < SEQ; kb += KTILE) {
    __syncthreads();
    // Cooperative K-tile stage: global fp32 -> LDS f16 (coalesced, 8 cols/thread)
    for (int idx = tid; idx < KTILE * (EMB / 8); idx += 256) {
      int r = idx / (EMB / 8);
      int c = (idx % (EMB / 8)) * 8;
      const float* src = xb + (size_t)(kb + r) * EMB + c;
      v8h h;
#pragma unroll
      for (int j = 0; j < 8; ++j) h[j] = (_Float16)src[j];
      *(v8h*)&ldsK[r * KPITCH + c] = h;
    }
    if (kb + KTILE < SEQ) {  // nudge next K tile toward L2
      __builtin_prefetch(xb + (size_t)(kb + KTILE + (tid >> 1)) * EMB + (tid & 1) * 384, 0, 1);
    }
    __syncthreads();

#pragma unroll
    for (int sb = 0; sb < KTILE / 16; ++sb) {
      const _Float16* kfrag = &ldsK[(sb * 16 + bcol) * KPITCH + bk16];

      // 3-deep software pipeline: keep 2 fragment-pairs in flight behind each WMMA
      HF16 kf[3];
#pragma unroll
      for (int pp = 0; pp < 2; ++pp) {
        kf[pp].h[0] = *(const v8h*)(kfrag + pp * 32);
        kf[pp].h[1] = *(const v8h*)(kfrag + pp * 32 + 8);
      }
      v8f c = {};
#pragma unroll
      for (int ks = 0; ks < NKSEG; ++ks) {
        if (ks + 2 < NKSEG) {
          HF16* nb = &kf[(ks + 2) % 3];
          nb->h[0] = *(const v8h*)(kfrag + (ks + 2) * 32);
          nb->h[1] = *(const v8h*)(kfrag + (ks + 2) * 32 + 8);
        }
        c = __builtin_amdgcn_wmma_f32_16x16x32_f16(false, qf[ks], false, kf[ks % 3].v,
                                                   (short)0, c, false, false);
      }

      // Online softmax update for this 16-key block.
      float2 v2 = ldsV[kb + sb * 16 + bcol];
      float smax[8];
#pragma unroll
      for (int r = 0; r < 8; ++r) smax[r] = c[r];
#pragma unroll
      for (int m = 8; m >= 1; m >>= 1)
#pragma unroll
        for (int r = 0; r < 8; ++r)
          smax[r] = fmaxf(smax[r], __shfl_xor(smax[r], m, 32));
#pragma unroll
      for (int r = 0; r < 8; ++r) {
        float nm   = fmaxf(runm[r], smax[r]);
        float corr = __expf(runm[r] - nm);
        runm[r] = nm;
        float p = __expf(c[r] - nm);
        lsum[r] = lsum[r] * corr + p;
        acc0[r] = acc0[r] * corr + p * v2.x;
        acc1[r] = acc1[r] * corr + p * v2.y;
      }
    }
  }

  // One final cross-lane sum per 16-lane group (deferred normalization).
#pragma unroll
  for (int m = 8; m >= 1; m >>= 1) {
#pragma unroll
    for (int r = 0; r < 8; ++r) {
      lsum[r] += __shfl_xor(lsum[r], m, 32);
      acc0[r] += __shfl_xor(acc0[r], m, 32);
      acc1[r] += __shfl_xor(acc1[r], m, 32);
    }
  }
  if ((lane & 15) == 0) {
    const float b0 = bias[0], b1 = bias[1];
    int mofs = (lane >> 4) * 8;  // lane 0 -> rows 0..7, lane 16 -> rows 8..15
#pragma unroll
    for (int r = 0; r < 8; ++r) {
      int   qr  = qbase + mofs + r;
      float inv = 1.0f / lsum[r];
      out[((size_t)b * SEQ + qr) * 2 + 0] = acc0[r] * inv + b0;
      out[((size_t)b * SEQ + qr) * 2 + 1] = acc1[r] * inv + b1;
    }
  }
}

extern "C" void kernel_launch(void* const* d_in, const int* in_sizes, int n_in,
                              void* d_out, int out_size, void* d_ws, size_t ws_size,
                              hipStream_t stream) {
  const float* x    = (const float*)d_in[0];  // [8, 2048, 768]
  const float* W    = (const float*)d_in[1];  // [2, 768]
  const float* bias = (const float*)d_in[2];  // [2]
  float* out = (float*)d_out;                 // [8, 2048, 2]
  float* vws = (float*)d_ws;                  // [8*2048, 2] scratch

  vproj_kernel<<<(BATCH * SEQ) / 8, 256, 0, stream>>>(x, W, vws);

  dim3 grid(SEQ / QTILE, BATCH);
  flash_attn_kernel<<<grid, 256, 0, stream>>>(x, vws, bias, out);
}
